// Self_Attn_78503412236502
// MI455X (gfx1250) — matrix operand order repaired and checked by reading, not compiled, not run on hardware
//
#include <hip/hip_runtime.h>

// MI455X / gfx1250, wave32. Flash-attention for the 1x1-conv self-attention block.
// Workspace (>= 22 MB):
//   Qh [B][N][32] f16 (2 MB) | Kh [B][N][32] f16 (2 MB) | Vh [B][C][N] f16 (16 MB)
//   Wh [320][256] f16 (160 KB) | scaleA[320] f32 | shiftA[320] f32

typedef _Float16 h16;
typedef __attribute__((ext_vector_type(16))) _Float16 v16h;
typedef __attribute__((ext_vector_type(8)))  _Float16 h8;    // 16-byte chunk
typedef __attribute__((ext_vector_type(8)))  float    v8f;
typedef __attribute__((ext_vector_type(4)))  unsigned int u32x4;
typedef __attribute__((ext_vector_type(4)))  int          i32x4;
typedef __attribute__((ext_vector_type(8)))  int          i32x8;

#define BB   8
#define CC   256
#define NN   4096
#define DQ   32
#define JC   64            // keys per chunk
#define NWV  8             // waves (query tiles) per block
#define OUTR 320           // 32 q + 32 k + 256 v output rows
#define XPITCH 264         // padded LDS pitch (halves) for the x^T tile
#define EPS  1e-5f

union V16 { v16h v; h8 h[2]; };

#ifndef __has_builtin
#define __has_builtin(x) 0
#endif
#if __has_builtin(__builtin_amdgcn_tensor_load_to_lds) && __has_builtin(__builtin_amdgcn_s_wait_tensorcnt)
#define USE_TDM 1
#else
#define USE_TDM 0
#endif

#if USE_TDM
// Tensor Data Mover: 2-D tile load (16-bit elements) per CDNA5 ISA D# layout.
__device__ __forceinline__ void tdm_load_2d(unsigned int lds_off, const h16* gaddr,
                                            unsigned int tdim0, unsigned int tdim1,
                                            unsigned int tile0, unsigned int tile1,
                                            unsigned int stride0)
{
    unsigned long long ga = (unsigned long long)(uintptr_t)gaddr;
    u32x4 g0;
    g0[0] = 1u;                                   // count=1, user descriptor
    g0[1] = lds_off;                              // LDS byte address
    g0[2] = (unsigned int)ga;                     // global addr [31:0]
    g0[3] = (unsigned int)((ga >> 32) & 0x01FFFFFFu) | (2u << 30);  // [56:32] | type=2
    i32x8 g1;
    g1[0] = (int)(1u << 16);                      // wg_mask=0, data_size=1 (2 bytes)
    g1[1] = (int)(tdim0 << 16);                   // tensor_dim0[15:0]
    g1[2] = (int)((tdim0 >> 16) | (tdim1 << 16)); // tensor_dim0[31:16] | tensor_dim1[15:0]
    g1[3] = (int)((tdim1 >> 16) | (tile0 << 16)); // tensor_dim1[31:16] | tile_dim0
    g1[4] = (int)tile1;                           // tile_dim1 | tile_dim2=0
    g1[5] = (int)stride0;                         // tensor_dim0_stride[31:0]
    g1[6] = 0;
    g1[7] = 0;
    i32x4 z4 = {};
#if __clang_major__ >= 23
    i32x8 z8 = {};
    __builtin_amdgcn_tensor_load_to_lds(g0, g1, z4, z4, z8, 0);
#else
    __builtin_amdgcn_tensor_load_to_lds(g0, g1, z4, z4, 0);
#endif
}
#endif

// ---------------------------------------------------------------------------
// Kernel 0: fold BN into per-row scale/shift and pack W = [wq;wk;wv] as f16.
// Grid: 320 blocks x 256 threads (block = output row).
// ---------------------------------------------------------------------------
__global__ __launch_bounds__(256) void prep_kernel(
    const float* __restrict__ wq, const float* __restrict__ bq, const float* __restrict__ gq,
    const float* __restrict__ betaq, const float* __restrict__ mq, const float* __restrict__ vq,
    const float* __restrict__ wk, const float* __restrict__ bk, const float* __restrict__ gk,
    const float* __restrict__ betak, const float* __restrict__ mk, const float* __restrict__ vk,
    const float* __restrict__ wv, const float* __restrict__ bv, const float* __restrict__ gv,
    const float* __restrict__ betav, const float* __restrict__ mv, const float* __restrict__ vv,
    h16* __restrict__ Wh, float* __restrict__ scaleA, float* __restrict__ shiftA)
{
    const int o = blockIdx.x;
    const int t = threadIdx.x;
    const float* src; float bias, mean, g, bet, var;
    if (o < 32)      { int oo = o;      src = wq + oo * CC; bias = bq[oo]; mean = mq[oo]; g = gq[oo]; bet = betaq[oo]; var = vq[oo]; }
    else if (o < 64) { int oo = o - 32; src = wk + oo * CC; bias = bk[oo]; mean = mk[oo]; g = gk[oo]; bet = betak[oo]; var = vk[oo]; }
    else             { int oo = o - 64; src = wv + oo * CC; bias = bv[oo]; mean = mv[oo]; g = gv[oo]; bet = betav[oo]; var = vv[oo]; }
    Wh[(size_t)o * CC + t] = (h16)src[t];
    if (t == 0) {
        float sc = g * rsqrtf(var + EPS);
        scaleA[o] = sc;
        shiftA[o] = (bias - mean) * sc + bet;
    }
}

// ---------------------------------------------------------------------------
// Kernel 1: WMMA projection. Y[320 x N] = W[320 x 256] * X[256 x N], then
// BN-affine + ReLU, f16 outputs in attention layouts.
// Per block: stage x[b][:, i0:i0+32] transposed into LDS as f16 xs[i][c]
// (padded pitch -> conflict-free B-fragment reads), then 8 waves x 5 tiles
// x 8 K-steps of v_wmma. D-layout gives each lane 8 *consecutive* output
// channels -> Q/K epilogue is a single 16-B store per lane.
// ---------------------------------------------------------------------------
__global__ __launch_bounds__(256) void proj_gemm_kernel(
    const float* __restrict__ x, const h16* __restrict__ Wh,
    const float* __restrict__ scaleA, const float* __restrict__ shiftA,
    h16* __restrict__ Qh, h16* __restrict__ Kh, h16* __restrict__ Vh)
{
    __shared__ h16 xs[32 * XPITCH];               // 16.5 KB, xs[i_local][c]
    const int t  = threadIdx.x;
    const int b  = blockIdx.x >> 7;               // 1024 blocks = 8 * 128
    const int i0 = (blockIdx.x & 127) * 32;
    const float* xb = x + (size_t)b * CC * NN;

    // Stage + transpose + f32->f16: coalesced float4 reads along i.
    #pragma unroll
    for (int it = 0; it < 8; ++it) {
        int id = it * 256 + t;
        int c = id >> 3, i4 = id & 7;
        float4 f = *(const float4*)(xb + (size_t)c * NN + i0 + i4 * 4);
        xs[(i4 * 4 + 0) * XPITCH + c] = (h16)f.x;
        xs[(i4 * 4 + 1) * XPITCH + c] = (h16)f.y;
        xs[(i4 * 4 + 2) * XPITCH + c] = (h16)f.z;
        xs[(i4 * 4 + 3) * XPITCH + c] = (h16)f.w;
    }
    __syncthreads();

    const int lane = t & 31;
    const int wid  = t >> 5;
    const int ln   = lane & 15;
    const int hh   = lane >> 4;

    for (int q = 0; q < 5; ++q) {
        int id = wid + 8 * q;                    // 0..39 output tiles
        int mt = id >> 1, nt = id & 1;
        int obase = mt * 16;

        v8f a8 = v8f{};
        #pragma unroll
        for (int kk = 0; kk < 8; ++kk) {
            V16 aw, bx;
            const h16* wrow = Wh + (size_t)(obase + ln) * CC + kk * 32;
            aw.h[0] = *(const h8*)(wrow + 8 * hh);
            aw.h[1] = *(const h8*)(wrow + 16 + 8 * hh);
            const h16* xrow = &xs[(nt * 16 + ln) * XPITCH + kk * 32 + 8 * hh];
            bx.h[0] = *(const h8*)(xrow);
            bx.h[1] = *(const h8*)(xrow + 16);
            a8 = __builtin_amdgcn_wmma_f32_16x16x32_f16(false, aw.v, false, bx.v, (short)0, a8, false, false);
        }

        int i     = i0 + nt * 16 + ln;
        int orow0 = obase + 8 * hh;              // 8 consecutive output rows per lane
        if (obase < 64) {                        // q / k branches: vectorized store
            h8 outv;
            #pragma unroll
            for (int r = 0; r < 8; ++r) {
                int o = orow0 + r;
                float y = a8[r] * scaleA[o] + shiftA[o];
                outv[r] = (h16)fmaxf(y, 0.f);
            }
            h16* dst = (obase < 32) ? (Qh + ((size_t)b * NN + i) * DQ + orow0)
                                    : (Kh + ((size_t)b * NN + i) * DQ + (orow0 - 32));
            *(h8*)dst = outv;
        } else {                                 // v branch: [C][N] layout
            #pragma unroll
            for (int r = 0; r < 8; ++r) {
                int o = orow0 + r;
                float y = a8[r] * scaleA[o] + shiftA[o];
                Vh[((size_t)b * CC + (o - 64)) * NN + i] = (h16)fmaxf(y, 0.f);
            }
        }
    }
}

// ---------------------------------------------------------------------------
// Kernel 2: flash attention. 256 blocks x 256 threads (8 waves). Each wave
// owns one 16-row query tile; 8 waves share TDM-staged K/V chunks (64 keys,
// double-buffered in dynamic LDS). Softmax row-sums via WMMA against ones.
// Dynamic LDS (88 KB):
//   Ks[2][64][32] (8 KB) | Vs[2][256][64] (64 KB) | Ps[8][16][64] (16 KB)
// ---------------------------------------------------------------------------
#define LDS_KS   0
#define LDS_VS   (2 * JC * DQ)                 // halves
#define LDS_PS   (LDS_VS + 2 * CC * JC)        // halves
#define LDS_HALVES (LDS_PS + NWV * 16 * JC)

__global__ __launch_bounds__(32 * NWV) void attn_kernel(
    const h16* __restrict__ Qh, const h16* __restrict__ Kh, const h16* __restrict__ Vh,
    const float* __restrict__ x, const float* __restrict__ gamma_p,
    float* __restrict__ out)
{
    extern __shared__ h16 smem[];
    h16* KsB = smem + LDS_KS;
    h16* VsB = smem + LDS_VS;
    h16* PsB = smem + LDS_PS;

    const int t    = threadIdx.x;
    const int lane = t & 31;
    const int wid  = t >> 5;                          // wave 0..7
    const int b    = blockIdx.x >> 5;                 // 256 blocks = 8 * 32
    const int i0   = ((blockIdx.x & 31) * NWV + wid) * 16;
    const int ln   = lane & 15;
    const int hh   = lane >> 4;

    // Q A-fragment (16x32): lane = row ln, K = 16h..16h+15
    V16 aq;
    {
        const h16* qrow = Qh + ((size_t)b * NN + i0 + ln) * DQ;
        aq.h[0] = *(const h8*)(qrow + 16 * hh);
        aq.h[1] = *(const h8*)(qrow + 16 * hh + 8);
    }

    // Constant ones B-fragment for WMMA row-sum.
    V16 vone;
    {
        h8 o8;
        #pragma unroll
        for (int i = 0; i < 8; ++i) o8[i] = (h16)1.f;
        vone.h[0] = o8; vone.h[1] = o8;
    }

    v8f acc[16];
    #pragma unroll
    for (int ct = 0; ct < 16; ++ct) acc[ct] = v8f{};
    float m_r[8], l_r[8];
    #pragma unroll
    for (int r = 0; r < 8; ++r) { m_r[r] = -1e30f; l_r[r] = 0.f; }

    const h16* Kb = Kh + (size_t)b * NN * DQ;
    const h16* Vb = Vh + (size_t)b * CC * NN;
    h16* Pw = PsB + wid * 16 * JC;

    const int NCH = NN / JC;    // 64 chunks

#if USE_TDM
    // Prologue: wave 0 kicks chunk 0 into buffer 0 via the Tensor Data Mover.
    if (wid == 0) {
        tdm_load_2d((unsigned int)(uintptr_t)(KsB), Kb, DQ, JC, DQ, JC, DQ);
        tdm_load_2d((unsigned int)(uintptr_t)(VsB), Vb, NN, CC, JC, CC, NN);
    }
#endif

    for (int tt = 0; tt < NCH; ++tt) {
        const int jb  = tt * JC;
        const int buf = tt & 1;
        __syncthreads();          // all consumers of buf^1 (iter tt-1) are done
#if USE_TDM
        if (wid == 0) {
            if (tt + 1 < NCH) {   // prefetch next chunk into the other buffer
                tdm_load_2d((unsigned int)(uintptr_t)(KsB + (buf ^ 1) * JC * DQ),
                            Kb + (size_t)(jb + JC) * DQ, DQ, JC, DQ, JC, DQ);
                tdm_load_2d((unsigned int)(uintptr_t)(VsB + (buf ^ 1) * CC * JC),
                            Vb + jb + JC, NN, CC, JC, CC, NN);
                __builtin_amdgcn_s_wait_tensorcnt(2);   // chunk tt complete
            } else {
                __builtin_amdgcn_s_wait_tensorcnt(0);
            }
        }
#else
        // Fallback staging: plain loads, all 256 threads.
        {
            int row = t >> 2, seg = t & 3;                 // K: 64 x 32 halves
            *(h8*)&KsB[buf * JC * DQ + row * DQ + seg * 8] =
                *(const h8*)(Kb + (size_t)(jb + row) * DQ + seg * 8);
        }
        #pragma unroll
        for (int it = 0; it < 8; ++it) {                   // V: 256 x 64 halves
            int id = it * 256 + t;
            int c = id >> 3, seg = id & 7;
            *(h8*)&VsB[buf * CC * JC + c * JC + seg * 8] =
                *(const h8*)(Vb + (size_t)c * NN + jb + seg * 8);
        }
#endif
        __syncthreads();

        const h16* Ksb = KsB + buf * JC * DQ;
        const h16* Vsb = VsB + buf * CC * JC;

        // S = Q K^T : four 16x16 tiles (B 32x16: lane = col ln, K = 16h..16h+15)
        v8f z = v8f{};
        v8f s0, s1, s2, s3;
        {
            V16 bk0, bk1, bk2, bk3;
            bk0.h[0] = *(const h8*)&Ksb[(0  + ln) * DQ + 16 * hh];
            bk0.h[1] = *(const h8*)&Ksb[(0  + ln) * DQ + 16 * hh + 8];
            bk1.h[0] = *(const h8*)&Ksb[(16 + ln) * DQ + 16 * hh];
            bk1.h[1] = *(const h8*)&Ksb[(16 + ln) * DQ + 16 * hh + 8];
            bk2.h[0] = *(const h8*)&Ksb[(32 + ln) * DQ + 16 * hh];
            bk2.h[1] = *(const h8*)&Ksb[(32 + ln) * DQ + 16 * hh + 8];
            bk3.h[0] = *(const h8*)&Ksb[(48 + ln) * DQ + 16 * hh];
            bk3.h[1] = *(const h8*)&Ksb[(48 + ln) * DQ + 16 * hh + 8];
            s0 = __builtin_amdgcn_wmma_f32_16x16x32_f16(false, aq.v, false, bk0.v, (short)0, z, false, false);
            s1 = __builtin_amdgcn_wmma_f32_16x16x32_f16(false, aq.v, false, bk1.v, (short)0, z, false, false);
            s2 = __builtin_amdgcn_wmma_f32_16x16x32_f16(false, aq.v, false, bk2.v, (short)0, z, false, false);
            s3 = __builtin_amdgcn_wmma_f32_16x16x32_f16(false, aq.v, false, bk3.v, (short)0, z, false, false);
        }

        // Online softmax: running max via 4-stage half-wave shuffle reduce;
        // P written straight to LDS in row-major f16 for A-fragment reload.
        float alpha_r[8];
        #pragma unroll
        for (int r = 0; r < 8; ++r) {
            float cmx = fmaxf(fmaxf(s0[r], s1[r]), fmaxf(s2[r], s3[r]));
            cmx = fmaxf(cmx, __shfl_xor(cmx, 1, 32));
            cmx = fmaxf(cmx, __shfl_xor(cmx, 2, 32));
            cmx = fmaxf(cmx, __shfl_xor(cmx, 4, 32));
            cmx = fmaxf(cmx, __shfl_xor(cmx, 8, 32));
            float mn = fmaxf(m_r[r], cmx);
            alpha_r[r] = __expf(m_r[r] - mn);
            m_r[r] = mn;
            h16* prow = Pw + (r + 8 * hh) * JC;
            prow[ln]      = (h16)__expf(s0[r] - mn);
            prow[16 + ln] = (h16)__expf(s1[r] - mn);
            prow[32 + ln] = (h16)__expf(s2[r] - mn);
            prow[48 + ln] = (h16)__expf(s3[r] - mn);
            #pragma unroll
            for (int ct = 0; ct < 16; ++ct) acc[ct][r] *= alpha_r[r];
        }

        // Same-wave DS pipe is in-order; wait + compiler fence for the RAW.
        asm volatile("s_wait_dscnt 0" ::: "memory");
        V16 ap0, ap1;      // P as A-fragments for j 0..31 and 32..63
        ap0.h[0] = *(const h8*)&Pw[ln * JC + 16 * hh];
        ap0.h[1] = *(const h8*)&Pw[ln * JC + 16 * hh + 8];
        ap1.h[0] = *(const h8*)&Pw[ln * JC + 32 + 16 * hh];
        ap1.h[1] = *(const h8*)&Pw[ln * JC + 32 + 16 * hh + 8];

        // Row-sum of P via WMMA against ones: D-layout replicates each row's
        // sum across its 16 lanes -- exactly the layout the l-update needs.
        v8f rs8 = __builtin_amdgcn_wmma_f32_16x16x32_f16(false, ap0.v, false, vone.v, (short)0, z, false, false);
        rs8     = __builtin_amdgcn_wmma_f32_16x16x32_f16(false, ap1.v, false, vone.v, (short)0, rs8, false, false);
        #pragma unroll
        for (int r = 0; r < 8; ++r) l_r[r] = l_r[r] * alpha_r[r] + rs8[r];

        // O += P (16 x 64j) * V^T (64j x 16c), 16 column tiles of C=256.
        #pragma unroll
        for (int ct = 0; ct < 16; ++ct) {
            const h16* vrow = Vsb + (ct * 16 + ln) * JC;
            V16 bv0, bv1;
            bv0.h[0] = *(const h8*)&vrow[16 * hh];
            bv0.h[1] = *(const h8*)&vrow[16 * hh + 8];
            bv1.h[0] = *(const h8*)&vrow[32 + 16 * hh];
            bv1.h[1] = *(const h8*)&vrow[32 + 16 * hh + 8];
            acc[ct] = __builtin_amdgcn_wmma_f32_16x16x32_f16(false, ap0.v, false, bv0.v, (short)0, acc[ct], false, false);
            acc[ct] = __builtin_amdgcn_wmma_f32_16x16x32_f16(false, ap1.v, false, bv1.v, (short)0, acc[ct], false, false);
        }
    }

    // Epilogue: out = gamma * (O / l) + x
    const float gamma = *gamma_p;
    float inv_l[8];
    #pragma unroll
    for (int r = 0; r < 8; ++r) inv_l[r] = 1.f / l_r[r];
    const float* xb = x   + (size_t)b * CC * NN;
    float*       ob = out + (size_t)b * CC * NN;
    #pragma unroll
    for (int ct = 0; ct < 16; ++ct) {
        int c = ct * 16 + ln;
        #pragma unroll
        for (int r = 0; r < 8; ++r) {
            int i = i0 + r + 8 * hh;
            size_t idx = (size_t)c * NN + i;
            ob[idx] = gamma * acc[ct][r] * inv_l[r] + xb[idx];
        }
    }
}

// ---------------------------------------------------------------------------
extern "C" void kernel_launch(void* const* d_in, const int* in_sizes, int n_in,
                              void* d_out, int out_size, void* d_ws, size_t ws_size,
                              hipStream_t stream) {
    const float* x     = (const float*)d_in[0];
    const float* wq    = (const float*)d_in[1];
    const float* bq    = (const float*)d_in[2];
    const float* gq    = (const float*)d_in[3];
    const float* betaq = (const float*)d_in[4];
    const float* mq    = (const float*)d_in[5];
    const float* vq    = (const float*)d_in[6];
    const float* wk    = (const float*)d_in[7];
    const float* bk    = (const float*)d_in[8];
    const float* gk    = (const float*)d_in[9];
    const float* betak = (const float*)d_in[10];
    const float* mk    = (const float*)d_in[11];
    const float* vk    = (const float*)d_in[12];
    const float* wv    = (const float*)d_in[13];
    const float* bv    = (const float*)d_in[14];
    const float* gv    = (const float*)d_in[15];
    const float* betav = (const float*)d_in[16];
    const float* mv    = (const float*)d_in[17];
    const float* vv    = (const float*)d_in[18];
    const float* gamma = (const float*)d_in[19];

    h16* Qh = (h16*)d_ws;
    h16* Kh = Qh + (size_t)BB * NN * DQ;
    h16* Vh = Kh + (size_t)BB * NN * DQ;
    h16* Wh = Vh + (size_t)BB * CC * NN;
    float* scaleA = (float*)(Wh + (size_t)OUTR * CC);
    float* shiftA = scaleA + OUTR;

    prep_kernel<<<OUTR, 256, 0, stream>>>(
        wq, bq, gq, betaq, mq, vq,
        wk, bk, gk, betak, mk, vk,
        wv, bv, gv, betav, mv, vv,
        Wh, scaleA, shiftA);

    proj_gemm_kernel<<<BB * (NN / 32), 256, 0, stream>>>(
        x, Wh, scaleA, shiftA, Qh, Kh, Vh);

    attn_kernel<<<BB * (NN / 16) / NWV, 32 * NWV, (size_t)LDS_HALVES * sizeof(h16), stream>>>(
        Qh, Kh, Vh, x, gamma, (float*)d_out);
}
